// SelfAttention_28561532518887
// MI455X (gfx1250) — matrix operand order repaired and checked by reading, not compiled, not run on hardware
//
#include <hip/hip_runtime.h>
#include <hip/hip_bf16.h>

// Problem constants (fixed by the reference harness)
#define DD   512
#define SS   2048
#define NB   8
#define ROWS (NB * SS)   // 16384
#define EPSV 1e-5f

typedef __attribute__((ext_vector_type(16))) _Float16 v16h;
typedef __attribute__((ext_vector_type(8)))  _Float16 v8h;
typedef __attribute__((ext_vector_type(8)))  float    v8f;

union F16Frag { v16h v; v8h h[2]; };
union H8 { v8h v; _Float16 e[8]; };

__device__ __forceinline__ v8f wmma16(v16h a, v16h b, v8f c) {
  // emits v_wmma_f32_16x16x32_f16
  return __builtin_amdgcn_wmma_f32_16x16x32_f16(
      /*neg_a=*/false, a, /*neg_b=*/false, b,
      /*c_mod=*/(short)0, c, /*reuse_a=*/false, /*reuse_b=*/false);
}

// ---------------------------------------------------------------------------
// Kernel 1: transpose + fp16-convert the two DxD weight matrices.
// WvhT[d][c] = f16(Wv[c][d]), W1hT[d][c] = f16(W1[c][d])
// ---------------------------------------------------------------------------
__global__ void prep_weights_kernel(const float* __restrict__ Wv,
                                    const float* __restrict__ W1,
                                    _Float16* __restrict__ WvhT,
                                    _Float16* __restrict__ W1hT) {
  int idx = blockIdx.x * 256 + threadIdx.x;   // 0 .. 512*512-1
  int d = idx >> 9;
  int c = idx & 511;
  WvhT[idx] = (_Float16)Wv[c * DD + d];
  W1hT[idx] = (_Float16)W1[c * DD + d];
}

// ---------------------------------------------------------------------------
// Kernel 2: per-row tiny projections (K=4) + x -> f16.
// queries[row][4], keysS[row][4] = (x@Wk+bk)*0.5, biasv[row] = x@Wb+bb
// One wave32 per row; lane-strided loads + shfl_xor reductions.
// ---------------------------------------------------------------------------
__global__ void rowstats_kernel(const float* __restrict__ x,
                                const float* __restrict__ Wk, const float* __restrict__ bk,
                                const float* __restrict__ Wq, const float* __restrict__ bq,
                                const float* __restrict__ Wb, const float* __restrict__ bb,
                                float* __restrict__ keysS, float* __restrict__ queries,
                                float* __restrict__ biasv, _Float16* __restrict__ xh) {
  const int wave = threadIdx.x >> 5;
  const int lane = threadIdx.x & 31;
  const int row  = blockIdx.x * 8 + wave;
  const float* xr = x + (size_t)row * DD;
  float a[9] = {0.f,0.f,0.f,0.f,0.f,0.f,0.f,0.f,0.f};
  #pragma unroll
  for (int kk = 0; kk < 16; ++kk) {
    int c = lane + 32 * kk;
    float xv = xr[c];
    xh[(size_t)row * DD + c] = (_Float16)xv;
    float4 wq = ((const float4*)Wq)[c];
    float4 wk = ((const float4*)Wk)[c];
    float  wb = Wb[c];
    a[0] += xv * wq.x; a[1] += xv * wq.y; a[2] += xv * wq.z; a[3] += xv * wq.w;
    a[4] += xv * wk.x; a[5] += xv * wk.y; a[6] += xv * wk.z; a[7] += xv * wk.w;
    a[8] += xv * wb;
  }
  #pragma unroll
  for (int i = 0; i < 9; ++i) {
    #pragma unroll
    for (int m = 16; m >= 1; m >>= 1) a[i] += __shfl_xor(a[i], m, 32);
  }
  if (lane == 0) {
    queries[row * 4 + 0] = a[0] + bq[0];
    queries[row * 4 + 1] = a[1] + bq[1];
    queries[row * 4 + 2] = a[2] + bq[2];
    queries[row * 4 + 3] = a[3] + bq[3];
    keysS[row * 4 + 0] = (a[4] + bk[0]) * 0.5f;   // fold 1/sqrt(K)
    keysS[row * 4 + 1] = (a[5] + bk[1]) * 0.5f;
    keysS[row * 4 + 2] = (a[6] + bk[2]) * 0.5f;
    keysS[row * 4 + 3] = (a[7] + bk[3]) * 0.5f;
    biasv[row] = a[8] + bb[0];
  }
}

// ---------------------------------------------------------------------------
// Kernel 3: values = x @ Wv + bv, stored TRANSPOSED as f16: vhT[b][d][j].
// Block = 32-row M tile; 8 waves each own a 64-wide N slice.
// Each wave: 2 M-subtiles x 4 N-tiles = 8 WMMA accumulators; one B fragment
// per N-tile per K-step feeds two WMMAs (2x B reuse).
// ---------------------------------------------------------------------------
__global__ void gemm_values_kernel(const _Float16* __restrict__ xh,
                                   const _Float16* __restrict__ WvhT,
                                   const float* __restrict__ bv,
                                   _Float16* __restrict__ vhT) {
  const int wave = threadIdx.x >> 5;
  const int lane = threadIdx.x & 31;
  const int i0   = blockIdx.x * 32;
  const int n0   = wave * 64;
  const int rsel = lane >> 4;      // 0: lanes 0-15, 1: lanes 16-31
  const int l16  = lane & 15;
  const int aoff = rsel * 8;
  v8f acc[2][4] = {};
  const _Float16* arow0 = xh + (size_t)(i0 + l16) * DD;
  const _Float16* arow1 = xh + (size_t)(i0 + 16 + l16) * DD;
  for (int c0 = 0; c0 < DD; c0 += 32) {
    F16Frag a0, a1;
    a0.h[0] = *(const v8h*)(arow0 + c0 + aoff);
    a0.h[1] = *(const v8h*)(arow0 + c0 + 16 + aoff);
    a1.h[0] = *(const v8h*)(arow1 + c0 + aoff);
    a1.h[1] = *(const v8h*)(arow1 + c0 + 16 + aoff);
    #pragma unroll
    for (int t = 0; t < 4; ++t) {
      int d = n0 + t * 16 + l16;
      F16Frag bfr;
      bfr.h[0] = *(const v8h*)(WvhT + (size_t)d * DD + c0 + aoff);
      bfr.h[1] = *(const v8h*)(WvhT + (size_t)d * DD + c0 + 16 + aoff);
      v16h bf = bfr.v;
      acc[0][t] = wmma16(a0.v, bf, acc[0][t]);
      acc[1][t] = wmma16(a1.v, bf, acc[1][t]);
    }
  }
  // Epilogue: +bv, convert to f16, store transposed (lane rows contiguous in j)
  const int b = i0 >> 11;                 // /2048
  #pragma unroll
  for (int ms = 0; ms < 2; ++ms) {
    const int jbase = (i0 & 2047) + ms * 16 + rsel * 8;
    #pragma unroll
    for (int t = 0; t < 4; ++t) {
      int d = n0 + t * 16 + l16;
      float bvd = bv[d];
      H8 o;
      #pragma unroll
      for (int r = 0; r < 8; ++r) o.e[r] = (_Float16)(acc[ms][t][r] + bvd);
      *(v8h*)(vhT + ((size_t)(b * DD + d)) * SS + jbase) = o.v;
    }
  }
}

// ---------------------------------------------------------------------------
// Kernel 4: fused sigmoid-attention + residual + LayerNorm1 (affine).
// Block = 32-row i tile of one batch. Streams j in chunks of 32:
//   score tile 32x32 (sigmoid(q.k+bias)*mask) -> LDS (double buffered)
//   -> two WMMA A fragments vs values B fragments from vhT (2x B reuse).
// Then res = LN(x + attention) -> res (f32) and resh (f16).
// ---------------------------------------------------------------------------
#define SA_STRIDE 40   // 32 + 8 pad (f16) to dodge LDS bank conflicts
__global__ void attention_kernel(const float* __restrict__ x,
                                 const float* __restrict__ mask,
                                 const float* __restrict__ keysS,
                                 const float* __restrict__ queries,
                                 const float* __restrict__ biasv,
                                 const _Float16* __restrict__ vhT,
                                 const float* __restrict__ gamma1,
                                 const float* __restrict__ beta1,
                                 float* __restrict__ res,
                                 _Float16* __restrict__ resh) {
  __shared__ _Float16 sA[2][32][SA_STRIDE];
  __shared__ float qTile[32][4];
  __shared__ float resT[32][DD + 4];

  const int tid  = threadIdx.x;
  const int wave = tid >> 5;
  const int lane = tid & 31;
  const int i0   = blockIdx.x * 32;          // tile start in [B*S]
  const int b    = i0 >> 11;
  const int n0   = wave * 64;
  const int rsel = lane >> 4;
  const int l16  = lane & 15;
  const int aoff = rsel * 8;

  if (tid < 128) ((float*)qTile)[tid] = queries[(size_t)i0 * 4 + tid];
  __syncthreads();

  // Per-thread score-producer coords: rows si, si+8, si+16, si+24; column sj
  const int si = tid >> 5;
  const int sj = tid & 31;
  float qr[4][4];
  #pragma unroll
  for (int p = 0; p < 4; ++p) {
    qr[p][0] = qTile[si + 8 * p][0];
    qr[p][1] = qTile[si + 8 * p][1];
    qr[p][2] = qTile[si + 8 * p][2];
    qr[p][3] = qTile[si + 8 * p][3];
  }

  const float* kb  = keysS + (size_t)b * SS * 4;
  const float* bsb = biasv + (size_t)b * SS;
  const float* mkb = mask  + (size_t)b * SS;

  v8f acc[2][4] = {};
  int buf = 0;
  for (int j0 = 0; j0 < SS; j0 += 32) {
    {   // produce 32x32 sigmoid score tile (4 entries/thread)
      int j = j0 + sj;
      float4 kv = ((const float4*)kb)[j];
      float bj = bsb[j];
      float mj = mkb[j];
      #pragma unroll
      for (int p = 0; p < 4; ++p) {
        float s = qr[p][0]*kv.x + qr[p][1]*kv.y + qr[p][2]*kv.z + qr[p][3]*kv.w + bj;
        s = mj / (1.f + __expf(-s));
        sA[buf][si + 8 * p][sj] = (_Float16)s;
      }
    }
    __syncthreads();
    F16Frag a0, a1;
    {
      const _Float16* ar0 = &sA[buf][l16][0];
      const _Float16* ar1 = &sA[buf][16 + l16][0];
      a0.h[0] = *(const v8h*)(ar0 + aoff);
      a0.h[1] = *(const v8h*)(ar0 + 16 + aoff);
      a1.h[0] = *(const v8h*)(ar1 + aoff);
      a1.h[1] = *(const v8h*)(ar1 + 16 + aoff);
    }
    if (j0 + 32 < SS)  // hint next values chunk into cache (global_prefetch_b8)
      __builtin_prefetch(vhT + ((size_t)(b * DD + n0 + l16)) * SS + j0 + 32, 0, 1);
    #pragma unroll
    for (int t = 0; t < 4; ++t) {
      int d = n0 + t * 16 + l16;
      F16Frag bfr;
      bfr.h[0] = *(const v8h*)(vhT + ((size_t)(b * DD + d)) * SS + j0 + aoff);
      bfr.h[1] = *(const v8h*)(vhT + ((size_t)(b * DD + d)) * SS + j0 + 16 + aoff);
      v16h bf = bfr.v;
      acc[0][t] = wmma16(a0.v, bf, acc[0][t]);
      acc[1][t] = wmma16(a1.v, bf, acc[1][t]);
    }
    buf ^= 1;
  }

  // residual into LDS tile
  #pragma unroll
  for (int ms = 0; ms < 2; ++ms) {
    #pragma unroll
    for (int t = 0; t < 4; ++t) {
      int d = n0 + t * 16 + l16;
      #pragma unroll
      for (int r = 0; r < 8; ++r) {
        int m = ms * 16 + rsel * 8 + r;
        resT[m][d] = acc[ms][t][r] + x[(size_t)(i0 + m) * DD + d];
      }
    }
  }
  __syncthreads();

  // LayerNorm1 (affine): each wave handles 4 rows
  #pragma unroll
  for (int rr = 0; rr < 4; ++rr) {
    int m = wave * 4 + rr;
    float s = 0.f, sq = 0.f;
    #pragma unroll
    for (int k = 0; k < 16; ++k) {
      float v = resT[m][lane + 32 * k];
      s += v; sq += v * v;
    }
    #pragma unroll
    for (int msk = 16; msk >= 1; msk >>= 1) {
      s  += __shfl_xor(s,  msk, 32);
      sq += __shfl_xor(sq, msk, 32);
    }
    float mu  = s * (1.f / DD);
    float var = sq * (1.f / DD) - mu * mu;
    float inv = rsqrtf(var + EPSV);
    size_t ro = (size_t)(i0 + m) * DD;
    #pragma unroll
    for (int k = 0; k < 16; ++k) {
      int c = lane + 32 * k;
      float y = (resT[m][c] - mu) * inv * gamma1[c] + beta1[c];
      res[ro + c]  = y;
      resh[ro + c] = (_Float16)y;
    }
  }
}

// ---------------------------------------------------------------------------
// Kernel 5: ff = relu(res @ W1 + b1); out = LN(res + ff) (no affine).
// Same M=32 register blocking as kernel 3.
// ---------------------------------------------------------------------------
__global__ void ffn_kernel(const float* __restrict__ res,
                           const _Float16* __restrict__ resh,
                           const _Float16* __restrict__ W1hT,
                           const float* __restrict__ b1,
                           float* __restrict__ out) {
  __shared__ float resT[32][DD + 4];
  const int wave = threadIdx.x >> 5;
  const int lane = threadIdx.x & 31;
  const int i0   = blockIdx.x * 32;
  const int n0   = wave * 64;
  const int rsel = lane >> 4;
  const int l16  = lane & 15;
  const int aoff = rsel * 8;
  v8f acc[2][4] = {};
  const _Float16* arow0 = resh + (size_t)(i0 + l16) * DD;
  const _Float16* arow1 = resh + (size_t)(i0 + 16 + l16) * DD;
  for (int c0 = 0; c0 < DD; c0 += 32) {
    F16Frag a0, a1;
    a0.h[0] = *(const v8h*)(arow0 + c0 + aoff);
    a0.h[1] = *(const v8h*)(arow0 + c0 + 16 + aoff);
    a1.h[0] = *(const v8h*)(arow1 + c0 + aoff);
    a1.h[1] = *(const v8h*)(arow1 + c0 + 16 + aoff);
    #pragma unroll
    for (int t = 0; t < 4; ++t) {
      int d = n0 + t * 16 + l16;
      F16Frag bfr;
      bfr.h[0] = *(const v8h*)(W1hT + (size_t)d * DD + c0 + aoff);
      bfr.h[1] = *(const v8h*)(W1hT + (size_t)d * DD + c0 + 16 + aoff);
      v16h bf = bfr.v;
      acc[0][t] = wmma16(a0.v, bf, acc[0][t]);
      acc[1][t] = wmma16(a1.v, bf, acc[1][t]);
    }
  }
  #pragma unroll
  for (int ms = 0; ms < 2; ++ms) {
    #pragma unroll
    for (int t = 0; t < 4; ++t) {
      int d = n0 + t * 16 + l16;
      float bd = b1[d];
      #pragma unroll
      for (int r = 0; r < 8; ++r) {
        int m = ms * 16 + rsel * 8 + r;
        float ff = acc[ms][t][r] + bd;
        ff = ff > 0.f ? ff : 0.f;
        resT[m][d] = ff + res[(size_t)(i0 + m) * DD + d];
      }
    }
  }
  __syncthreads();
  #pragma unroll
  for (int rr = 0; rr < 4; ++rr) {
    int m = wave * 4 + rr;
    float s = 0.f, sq = 0.f;
    #pragma unroll
    for (int k = 0; k < 16; ++k) {
      float v = resT[m][lane + 32 * k];
      s += v; sq += v * v;
    }
    #pragma unroll
    for (int msk = 16; msk >= 1; msk >>= 1) {
      s  += __shfl_xor(s,  msk, 32);
      sq += __shfl_xor(sq, msk, 32);
    }
    float mu  = s * (1.f / DD);
    float var = sq * (1.f / DD) - mu * mu;
    float inv = rsqrtf(var + EPSV);
    size_t ro = (size_t)(i0 + m) * DD;
    #pragma unroll
    for (int k = 0; k < 16; ++k) {
      int c = lane + 32 * k;
      out[ro + c] = (resT[m][c] - mu) * inv;
    }
  }
}

// ---------------------------------------------------------------------------
extern "C" void kernel_launch(void* const* d_in, const int* in_sizes, int n_in,
                              void* d_out, int out_size, void* d_ws, size_t ws_size,
                              hipStream_t stream) {
  (void)in_sizes; (void)n_in; (void)out_size; (void)ws_size;
  const float* x      = (const float*)d_in[0];
  const float* mask   = (const float*)d_in[1];
  const float* Wk     = (const float*)d_in[2];
  const float* bk     = (const float*)d_in[3];
  const float* Wq     = (const float*)d_in[4];
  const float* bq     = (const float*)d_in[5];
  const float* Wv     = (const float*)d_in[6];
  const float* bv     = (const float*)d_in[7];
  const float* Wb     = (const float*)d_in[8];
  const float* bb     = (const float*)d_in[9];
  const float* W1     = (const float*)d_in[10];
  const float* b1     = (const float*)d_in[11];
  const float* gamma1 = (const float*)d_in[12];
  const float* beta1  = (const float*)d_in[13];
  float* out = (float*)d_out;

  // Workspace carve-out (~82 MB total)
  char* p = (char*)d_ws;
  auto take = [&p](size_t bytes) -> char* {
    char* r = p;
    p += (bytes + 255) & ~(size_t)255;
    return r;
  };
  _Float16* xh      = (_Float16*)take((size_t)ROWS * DD * 2);
  _Float16* resh    = (_Float16*)take((size_t)ROWS * DD * 2);
  _Float16* vhT     = (_Float16*)take((size_t)NB * DD * SS * 2);
  _Float16* WvhT    = (_Float16*)take((size_t)DD * DD * 2);
  _Float16* W1hT    = (_Float16*)take((size_t)DD * DD * 2);
  float*    keysS   = (float*)take((size_t)ROWS * 4 * 4);
  float*    queries = (float*)take((size_t)ROWS * 4 * 4);
  float*    biasv   = (float*)take((size_t)ROWS * 4);
  float*    res     = (float*)take((size_t)ROWS * DD * 4);

  prep_weights_kernel<<<dim3((DD * DD) / 256), 256, 0, stream>>>(Wv, W1, WvhT, W1hT);
  rowstats_kernel<<<dim3(ROWS / 8), 256, 0, stream>>>(x, Wk, bk, Wq, bq, Wb, bb,
                                                      keysS, queries, biasv, xh);
  gemm_values_kernel<<<dim3(ROWS / 32), 256, 0, stream>>>(xh, WvhT, bv, vhT);
  attention_kernel<<<dim3(ROWS / 32), 256, 0, stream>>>(x, mask, keysS, queries, biasv,
                                                        vhT, gamma1, beta1, res, resh);
  ffn_kernel<<<dim3(ROWS / 32), 256, 0, stream>>>(res, resh, W1hT, b1, out);
}
